// LambdaLayer_25443386261929
// MI455X (gfx1250) — matrix operand order repaired and checked by reading, not compiled, not run on hardware
//
#include <hip/hip_runtime.h>
#include <stdint.h>

// ---------------------------------------------------------------------------
// LambdaLayer on gfx1250 (MI455X). Dominant cost: per-n GEMM
//   lambdap[n] (16 x 768) = posT[n] (16 x 1024, bf16) @ v_bf (1024 x 768, bf16)
// done with v_wmma_f32_16x16x32_bf16; A panel staged in LDS with the Tensor
// Data Mover (tensor_load_to_lds, TENSORcnt) or async-to-LDS b128 fallback.
// ---------------------------------------------------------------------------

typedef unsigned short u16;
typedef __bf16 bf16;
typedef __attribute__((ext_vector_type(16))) __bf16 bf16x16;
typedef __attribute__((ext_vector_type(8)))  float  f32x8;
typedef __attribute__((ext_vector_type(4)))  int    v4i;
typedef __attribute__((ext_vector_type(4)))  unsigned int v4u;
typedef __attribute__((ext_vector_type(8)))  int    v8i;

#define BQ   8
#define CCH  384
#define NPIX 1024
#define KEYS 16
#define HEADS 4
#define VSZ  96
#define ZROWS 176       // 64 q + 16 k + 96 v

// ---- CDNA5 feature probes ---------------------------------------------------
#if defined(__has_builtin)
#  if __has_builtin(__builtin_amdgcn_global_load_async_to_lds_b128)
#    define USE_ASYNC_LDS 1
#  endif
#  if __has_builtin(__builtin_amdgcn_tensor_load_to_lds) && \
      __has_builtin(__builtin_amdgcn_s_wait_tensorcnt)
#    define USE_TDM 1
#  endif
#endif
#ifndef USE_ASYNC_LDS
#  define USE_ASYNC_LDS 0
#endif
#ifndef USE_TDM
#  define USE_TDM 0
#endif

__device__ __forceinline__ void async_copy16(const void* g, void* l) {
#if USE_ASYNC_LDS
  __builtin_amdgcn_global_load_async_to_lds_b128(
      (__attribute__((address_space(1))) v4i*)g,
      (__attribute__((address_space(3))) v4i*)l, 0, 0);
#else
  *(uint4*)l = *(const uint4*)g;
#endif
}

__device__ __forceinline__ void wait_async_zero() {
#if USE_ASYNC_LDS
#  if __has_builtin(__builtin_amdgcn_s_wait_asynccnt)
  __builtin_amdgcn_s_wait_asynccnt(0);
#  else
  asm volatile("s_wait_asynccnt 0" ::: "memory");
#  endif
#endif
}

// ---- helpers ---------------------------------------------------------------
__device__ __forceinline__ u16 f2bf(float f) {   // RNE float->bf16 bits
  unsigned int u = __float_as_uint(f);
  unsigned int r = u + 0x7fffu + ((u >> 16) & 1u);
  return (u16)(r >> 16);
}

union Frag {          // 32 bytes: two b128 loads <-> one WMMA 16-bit operand
  uint4   u[2];
  bf16x16 v;
};

__device__ __forceinline__ f32x8 wmma_bf16(bf16x16 a, bf16x16 b, f32x8 c) {
  // D(f32 16x16) = A(bf16 16x32) * B(bf16 32x16) + C
  return __builtin_amdgcn_wmma_f32_16x16x32_bf16(
      false, a, false, b, (short)0, c, false, false);
}

// A fragment (16x32 bf16, M=row, K contiguous in memory):
//   lane L holds row M = L%16; elems 0..7 -> K = k0 + (L/16)*8,
//   elems 8..15 -> K = k0 + 16 + (L/16)*8   (ISA 7.12.2 16-bit A layout)
__device__ __forceinline__ Frag load_a(const u16* base /*row ptr + sel*8*/, int k0) {
  Frag f;
  const u16* p = base + k0;
  f.u[0] = *(const uint4*)(p);
  f.u[1] = *(const uint4*)(p + 16);
  return f;
}
// B fragment (32x16 bf16, columns contiguous over K in memory):
//   lane L holds column N = L%16; elems 0..7 -> K = k0 + (L/16)*8,
//   elems 8..15 -> K = k0 + 16 + (L/16)*8 (same order as A)
__device__ __forceinline__ Frag load_b(const u16* base /*col ptr + sel*8*/, int k0) {
  Frag f;
  const u16* p = base + k0;
  f.u[0] = *(const uint4*)(p);
  f.u[1] = *(const uint4*)(p + 16);
  return f;
}

// ---------------------------------------------------------------------------
// K0a: Wcat bf16 [176][384]  (rows 0..63=Wq, 64..79=Wk, 80..175=Wv)
__global__ void wcat_kernel(const float* __restrict__ Wq, const float* __restrict__ Wk,
                            const float* __restrict__ Wv, u16* __restrict__ wcat) {
  int i = blockIdx.x * 256 + threadIdx.x;          // < 176*384 = 67584
  int row = i / CCH, c = i % CCH;
  float v;
  if (row < 64)      v = Wq[row * CCH + c];
  else if (row < 80) v = Wk[(row - 64) * CCH + c];
  else               v = Wv[(row - 80) * CCH + c];
  wcat[i] = f2bf(v);
}

// K0b: x (B,C,N) f32 -> xbT (B,N,C) bf16 via LDS tile transpose
__global__ void xT_kernel(const float* __restrict__ x, u16* __restrict__ xbT) {
  __shared__ float tile[32][33];
  int b = blockIdx.z, c0 = blockIdx.y * 32, n0 = blockIdx.x * 32;
  int tx = threadIdx.x, ty = threadIdx.y;
  for (int j = ty; j < 32; j += 8)
    tile[tx][j] = x[((b * CCH) + (c0 + j)) * NPIX + n0 + tx];
  __syncthreads();
  for (int j = ty; j < 32; j += 8)
    xbT[((b * NPIX) + (n0 + j)) * CCH + c0 + tx] = f2bf(tile[j][tx]);
}

// K0c: pos (N,M,16) f32 -> posT (N,16,M) bf16
__global__ void posT_kernel(const float* __restrict__ pos, u16* __restrict__ posT) {
  __shared__ float tile[1024];
  int n = blockIdx.y, m0 = blockIdx.x * 64;
  int t = threadIdx.x;
  for (int i = 0; i < 4; ++i) {
    int idx = i * 256 + t;                  // 0..1023 (coalesced read)
    int ml = idx >> 4, kk = idx & 15;
    tile[kk * 64 + ml] = pos[((n * NPIX) + (m0 + ml)) * 16 + kk];
  }
  __syncthreads();
  for (int i = 0; i < 4; ++i) {
    int o = i * 256 + t;                    // linear = kk*64+ml (coalesced write)
    int kk = o >> 6, ml = o & 63;
    posT[((n * 16) + kk) * NPIX + m0 + ml] = f2bf(tile[o]);
  }
}

// K1: zcat[b][176][1024] = Wcat @ x[b]   (WMMA bf16, K=384)
__global__ __launch_bounds__(256) void proj_kernel(const u16* __restrict__ wcat,
                                                   const u16* __restrict__ xbT,
                                                   float* __restrict__ zcat) {
  const int t = threadIdx.x, lane = t & 31, w = t >> 5;
  const int gt = blockIdx.x * 8 + w;               // 0..5631
  const int b = gt / 704, rem = gt % 704;
  const int mt = rem >> 6, nt = rem & 63;
  const int col = lane & 15, sel = lane >> 4;
  const u16* aptr = wcat + (mt * 16 + col) * CCH + sel * 8;
  const u16* bptr = xbT + ((b * NPIX) + nt * 16 + col) * CCH + sel * 8;
  f32x8 acc = {0.f, 0.f, 0.f, 0.f, 0.f, 0.f, 0.f, 0.f};
  for (int ks = 0; ks < CCH / 32; ++ks) {
    Frag a = load_a(aptr, ks * 32);
    Frag bb = load_b(bptr, ks * 32);
    acc = wmma_bf16(a.v, bb.v, acc);
  }
#pragma unroll
  for (int r = 0; r < 8; ++r) {
    int row = mt * 16 + r + sel * 8;               // C layout: lanes16-31 -> M=r+8
    zcat[((b * ZROWS) + row) * NPIX + nt * 16 + col] = acc[r];
  }
}

// K2: BatchNorm (training-mode biased stats over B*N) on q rows + v rows.
//     q normalized in place (fp32); v written as bf16 [b][vv][m].
__global__ void bn_kernel(const float* __restrict__ gq, const float* __restrict__ bq,
                          const float* __restrict__ gv, const float* __restrict__ bv,
                          float* __restrict__ zcat, u16* __restrict__ vbf) {
  int ch = blockIdx.x;                              // 0..159
  int zrow = (ch < 64) ? ch : ch + 16;
  float g = (ch < 64) ? gq[ch] : gv[ch - 64];
  float be = (ch < 64) ? bq[ch] : bv[ch - 64];
  __shared__ float s1[256], s2[256];
  int t = threadIdx.x;
  float sum = 0.f, sq = 0.f;
  for (int i = t; i < BQ * NPIX; i += 256) {
    int b = i >> 10, n = i & 1023;
    float v = zcat[((b * ZROWS) + zrow) * NPIX + n];
    sum += v; sq += v * v;
  }
  s1[t] = sum; s2[t] = sq; __syncthreads();
  for (int s = 128; s > 0; s >>= 1) {
    if (t < s) { s1[t] += s1[t + s]; s2[t] += s2[t + s]; }
    __syncthreads();
  }
  float mean = s1[0] * (1.f / (BQ * NPIX));
  float var = s2[0] * (1.f / (BQ * NPIX)) - mean * mean;
  float rstd = rsqrtf(var + 1e-5f);
  for (int i = t; i < BQ * NPIX; i += 256) {
    int b = i >> 10, n = i & 1023;
    float v = zcat[((b * ZROWS) + zrow) * NPIX + n];
    float o = (v - mean) * rstd * g + be;
    if (ch < 64) zcat[((b * ZROWS) + zrow) * NPIX + n] = o;
    else         vbf[((b * VSZ) + (ch - 64)) * NPIX + n] = f2bf(o);
  }
}

// K3: softmax over N on k rows; write bf16 [b][k][m]
__global__ void softmax_kernel(const float* __restrict__ zcat, u16* __restrict__ kbf) {
  int row = blockIdx.x;                             // b*16 + kk
  int b = row >> 4, kk = row & 15;
  const float* src = zcat + ((b * ZROWS) + 64 + kk) * NPIX;
  __shared__ float sred[256];
  int t = threadIdx.x;
  float mx = -3.4e38f;
  for (int i = t; i < NPIX; i += 256) mx = fmaxf(mx, src[i]);
  sred[t] = mx; __syncthreads();
  for (int s = 128; s > 0; s >>= 1) {
    if (t < s) sred[t] = fmaxf(sred[t], sred[t + s]);
    __syncthreads();
  }
  mx = sred[0]; __syncthreads();
  float sum = 0.f;
  for (int i = t; i < NPIX; i += 256) sum += __expf(src[i] - mx);
  sred[t] = sum; __syncthreads();
  for (int s = 128; s > 0; s >>= 1) {
    if (t < s) sred[t] += sred[t + s];
    __syncthreads();
  }
  float inv = 1.f / sred[0];
  for (int i = t; i < NPIX; i += 256)
    kbf[row * NPIX + i] = f2bf(__expf(src[i] - mx) * inv);
}

// K4: lambdac[b][16][96] = ksoft(16x1024) @ v(96x1024)^T  (WMMA bf16)
__global__ __launch_bounds__(256) void lamc_kernel(const u16* __restrict__ kbf,
                                                   const u16* __restrict__ vbf,
                                                   float* __restrict__ lamc) {
  const int b = blockIdx.x;
  const int t = threadIdx.x, lane = t & 31, w = t >> 5;
  if (w >= 6) return;                               // wave-uniform; EXEC all-1 in live waves
  const int col = lane & 15, sel = lane >> 4;
  const int vv0 = w * 16;
  const u16* aptr = kbf + ((b * KEYS) + col) * NPIX + sel * 8;
  const u16* bptr = vbf + ((b * VSZ) + vv0 + col) * NPIX + sel * 8;
  f32x8 acc = {0.f, 0.f, 0.f, 0.f, 0.f, 0.f, 0.f, 0.f};
  for (int ks = 0; ks < NPIX / 32; ++ks) {
    Frag a = load_a(aptr, ks * 32);
    Frag bb = load_b(bptr, ks * 32);
    acc = wmma_bf16(a.v, bb.v, acc);
  }
#pragma unroll
  for (int r = 0; r < 8; ++r)
    lamc[((b * KEYS) + (r + sel * 8)) * VSZ + vv0 + col] = acc[r];
}

// K5 (dominant): per n, lam(16x768) = posT[n] @ v_bf; epilogue fuses +lambdac
// and the q-contraction over k.  YT[n][b*384 + h*96 + vv] (n-contiguous writes).
__global__ __launch_bounds__(256) void lambda_main_kernel(
    const u16* __restrict__ posT, const u16* __restrict__ vbf,
    const float* __restrict__ zcat, const float* __restrict__ lamc,
    float* __restrict__ YT) {
  __shared__ u16 sA[16 * NPIX];                     // 32 KB A panel for this n
  const int n = blockIdx.x;
  const int t = threadIdx.x;
  const int lane = t & 31, w = t >> 5;
  const int col = lane & 15, sel = lane >> 4;
  const u16* src = posT + n * (16 * NPIX);

#if USE_TDM
  // One TDM descriptor per workgroup: 1-D tile of 16384 bf16 (32 KB),
  // posT[n] -> sA.  D# fields per ISA 8.3/8.4; issued by wave 0 only
  // (TDM ignores EXEC; one instruction = one DMA), tracked on TENSORcnt.
  if (w == 0) {
    unsigned long long ga =
        (unsigned long long)(__attribute__((address_space(1))) const char*)src;
    unsigned lds_off =
        (unsigned)(unsigned long long)(__attribute__((address_space(3))) u16*)sA;
    v4u g0;
    g0.x = 1u;                                        // count=1, user D#
    g0.y = lds_off;                                   // lds_addr [63:32]
    g0.z = (unsigned)(ga & 0xffffffffu);              // global_addr [95:64]
    g0.w = (unsigned)((ga >> 32) & 0x01ffffffu)       // global_addr [120:96]
         | (2u << 30);                                // type=2 ("image") [127:126]
    v8i g1;
    g1[0] = (int)(1u << 16);                          // data_size=1 (2 bytes)
    g1[1] = (int)(16384u << 16);                      // tensor_dim0[15:0] @ [63:48]
    g1[2] = (int)(1u << 16);                          // tensor_dim0[31:16]=0; tensor_dim1=1
    g1[3] = (int)(16384u << 16);                      // tile_dim0=16384 @ [127:112]
    g1[4] = 0;                                        // tile_dim1=0, tile_dim2=0 (unused)
    g1[5] = 16384;                                    // tensor_dim0_stride[31:0]
    g1[6] = 0;                                        // stride0[47:32]=0, stride1 lo=0
    g1[7] = 0;
    v4i gz4 = {0, 0, 0, 0};                           // groups 2/3 unused (<=2D)
    v8i gz8 = {0, 0, 0, 0, 0, 0, 0, 0};               // extra group (6-arg variant)
    __builtin_amdgcn_tensor_load_to_lds(g0, g1, gz4, gz4, gz8, 0);
    __builtin_amdgcn_s_wait_tensorcnt(0);
  }
#else
#pragma unroll
  for (int i = 0; i < 8; ++i) {
    int off = (t + i * 256) * 8;                      // 16-byte chunks
    async_copy16(src + off, sA + off);
  }
  wait_async_zero();
#endif
  __syncthreads();

  const u16* aptr = sA + col * NPIX + sel * 8;        // A row = k_idx = col

  f32x8 acc[6];
#pragma unroll
  for (int i = 0; i < 6; ++i) acc[i] = (f32x8){0.f,0.f,0.f,0.f,0.f,0.f,0.f,0.f};

  const u16* bptr[6];
#pragma unroll
  for (int i = 0; i < 6; ++i) {
    int tile = w * 6 + i;                             // 0..47; b = tile/6, vv-tile = tile%6
    int b = tile / 6, vv0 = (tile % 6) * 16;
    bptr[i] = vbf + ((b * VSZ) + vv0 + col) * NPIX + sel * 8;
  }

  for (int ks = 0; ks < NPIX / 32; ++ks) {            // K = m, 32 per WMMA
    __builtin_prefetch(bptr[0] + (ks + 8) * 32, 0, 0);// stream-ahead hint on B
    Frag a = load_a(aptr, ks * 32);                   // LDS ds_load_b128 x2, reused 6x
#pragma unroll
    for (int i = 0; i < 6; ++i) {
      Frag bb = load_b(bptr[i], ks * 32);             // global b128 x2 (v_bf hot in L2)
      acc[i] = wmma_bf16(a.v, bb.v, acc[i]);
    }
  }

#pragma unroll
  for (int i = 0; i < 6; ++i) {
    int tile = w * 6 + i;
    int b = tile / 6, vv = (tile % 6) * 16 + col;
    const float* lc = lamc + (b * KEYS) * VSZ + vv;
    float lam[8];
#pragma unroll
    for (int r = 0; r < 8; ++r) {
      int k_idx = r + sel * 8;                        // C layout: lanes16-31 -> M=r+8
      lam[r] = acc[i][r] + lc[k_idx * VSZ];           // + lambdac
    }
#pragma unroll
    for (int h = 0; h < 4; ++h) {
      float p = 0.f;
#pragma unroll
      for (int r = 0; r < 8; ++r) {
        int k_idx = r + sel * 8;
        p += zcat[((b * ZROWS) + (h * KEYS + k_idx)) * NPIX + n] * lam[r];
      }
      p += __shfl_xor(p, 16, 32);                     // combine k halves across lane halves
      if (sel == 0)
        YT[(size_t)n * (BQ * CCH) + b * CCH + h * VSZ + vv] = p;
    }
  }
}

// K6: YT (1024 x 3072) -> out (3072 x 1024), coalesced both ways
__global__ void outT_kernel(const float* __restrict__ YT, float* __restrict__ out) {
  __shared__ float tile[32][33];
  int c0 = blockIdx.y * 32, n0 = blockIdx.x * 32;
  int tx = threadIdx.x, ty = threadIdx.y;
  for (int j = ty; j < 32; j += 8)
    tile[j][tx] = YT[(size_t)(n0 + j) * (BQ * CCH) + c0 + tx];
  __syncthreads();
  for (int j = ty; j < 32; j += 8)
    out[(size_t)(c0 + j) * NPIX + n0 + tx] = tile[tx][j];
}

// ---------------------------------------------------------------------------
extern "C" void kernel_launch(void* const* d_in, const int* in_sizes, int n_in,
                              void* d_out, int out_size, void* d_ws, size_t ws_size,
                              hipStream_t stream) {
  (void)in_sizes; (void)n_in; (void)out_size; (void)ws_size;
  const float* x   = (const float*)d_in[0];
  const float* Wq  = (const float*)d_in[1];
  const float* Wk  = (const float*)d_in[2];
  const float* Wv  = (const float*)d_in[3];
  const float* gq  = (const float*)d_in[4];
  const float* bq  = (const float*)d_in[5];
  const float* gv  = (const float*)d_in[6];
  const float* bv  = (const float*)d_in[7];
  const float* pos = (const float*)d_in[8];
  float* out = (float*)d_out;

  char* ws = (char*)d_ws;
  size_t off = 0;
  float* zcat = (float*)(ws + off); off += (size_t)BQ * ZROWS * NPIX * 4;   // 5.77 MB
  u16* xbT    = (u16*)(ws + off);   off += (size_t)BQ * NPIX * CCH * 2;     // 6.29 MB
  u16* wcat   = (u16*)(ws + off);   off += (size_t)ZROWS * CCH * 2;         // 132 KB
  u16* posT   = (u16*)(ws + off);   off += (size_t)NPIX * 16 * NPIX * 2;    // 32 MB
  u16* vbf    = (u16*)(ws + off);   off += (size_t)BQ * VSZ * NPIX * 2;     // 1.5 MB
  u16* kbf    = (u16*)(ws + off);   off += (size_t)BQ * KEYS * NPIX * 2;    // 256 KB
  float* lamc = (float*)(ws + off); off += (size_t)BQ * KEYS * VSZ * 4;     // 48 KB
  float* YT   = (float*)(ws + off); off += (size_t)NPIX * BQ * CCH * 4;     // 12.6 MB

  wcat_kernel<<<(ZROWS * CCH) / 256, 256, 0, stream>>>(Wq, Wk, Wv, wcat);
  xT_kernel<<<dim3(NPIX / 32, CCH / 32, BQ), dim3(32, 8), 0, stream>>>(x, xbT);
  posT_kernel<<<dim3(NPIX / 64, NPIX), 256, 0, stream>>>(pos, posT);
  proj_kernel<<<(BQ * 11 * 64) / 8, 256, 0, stream>>>(wcat, xbT, zcat);
  bn_kernel<<<160, 256, 0, stream>>>(gq, bq, gv, bv, zcat, vbf);
  softmax_kernel<<<BQ * KEYS, 256, 0, stream>>>(zcat, kbf);
  lamc_kernel<<<BQ, 256, 0, stream>>>(kbf, vbf, lamc);
  lambda_main_kernel<<<NPIX, 256, 0, stream>>>(posT, vbf, zcat, lamc, YT);
  outT_kernel<<<dim3(NPIX / 32, (BQ * CCH) / 32), dim3(32, 8), 0, stream>>>(YT, out);
}
